// MambaBlock_78683800862770
// MI455X (gfx1250) — compile-verified, matrix operand order checked by reading, not run on hardware
//
#include <hip/hip_runtime.h>
#include <hip/hip_bf16.h>

typedef __attribute__((ext_vector_type(16))) _Float16 v16h;
typedef __attribute__((ext_vector_type(8)))  _Float16 v8h;
typedef __attribute__((ext_vector_type(8)))  float    v8f;
typedef __attribute__((ext_vector_type(4)))  int      v4i;

#define D_MODEL 1024
#define D_STATE 16
#define D_CONV  4
#define D_INNER 2048
#define BATCH   4
#define SEQ     4096
#define NROWS   (BATCH * SEQ)   // 16384

#define AS1 __attribute__((address_space(1)))
#define AS3 __attribute__((address_space(3)))

#if __has_builtin(__builtin_amdgcn_global_load_async_to_lds_b128) && \
    __has_builtin(__builtin_amdgcn_s_wait_asynccnt)
#define USE_ASYNC_LDS 1
#else
#define USE_ASYNC_LDS 0
#endif

// LDS tile geometry: 128 rows x 32 halves, padded to 40 halves/row (80 B) to
// spread the 64 LDS banks under ds_load_b128 (gcd(20,64)=4 -> 16 distinct
// bank groups across the 16 row-lanes).
#define LDS_ROW 40

template <int N>
__device__ __forceinline__ void wait_async() {
#if USE_ASYNC_LDS
    __builtin_amdgcn_s_wait_asynccnt(N);
#endif
}

// Load one 16x32 f16 WMMA A/B fragment from an LDS tile (row-major, LDS_ROW
// halves per row). Per CDNA5 ISA layout: lanes 0-15 hold row m = lane,
// K = {0..7, 16..23}; lanes 16-31 hold row m = lane-16 with K offset +8.
__device__ __forceinline__ v16h load_frag_lds(const _Float16* __restrict__ p) {
    v8h lo = *(const v8h*)p;          // K = uo .. uo+7
    v8h hi = *(const v8h*)(p + 16);   // K = uo+16 .. uo+23
    v16h r;
#pragma unroll
    for (int i = 0; i < 8; ++i) { r[i] = lo[i]; r[i + 8] = hi[i]; }
    return r;
}

__device__ __forceinline__ float silu_f(float v) {
    return v / (1.0f + __expf(-v));
}

// ---------------------------------------------------------------------------
// Small streaming kernels
// ---------------------------------------------------------------------------

__global__ void cast_f32_to_f16_kernel(const float* __restrict__ src,
                                       _Float16* __restrict__ dst, int n) {
    int i = blockIdx.x * blockDim.x + threadIdx.x;
    if (i < n) dst[i] = (_Float16)src[i];
}

// cw[k] = (1/16) * sum_{n=17..32} xproj_w[n][k]  (only mean(C) is ever used)
__global__ void cw_kernel(const float* __restrict__ xproj_w,
                          float* __restrict__ cw) {
    int k = blockIdx.x * blockDim.x + threadIdx.x;
    if (k < D_INNER) {
        float s = 0.0f;
#pragma unroll
        for (int j = 0; j < D_STATE; ++j)
            s += xproj_w[(size_t)(1 + D_STATE + j) * D_INNER + k];
        cw[k] = s * (1.0f / (float)D_STATE);
    }
}

// LayerNorm over D_MODEL=1024, one wave32 per row, output f16.
__global__ __launch_bounds__(256) void ln_kernel(const float* __restrict__ x,
                                                 const float* __restrict__ w,
                                                 const float* __restrict__ b,
                                                 _Float16* __restrict__ xn) {
    int wave = threadIdx.x >> 5;
    int lane = threadIdx.x & 31;
    int row  = blockIdx.x * 8 + wave;
    if (row >= NROWS) return;
    const float* xr = x + (size_t)row * D_MODEL;

    float v[32];
    float s = 0.0f;
#pragma unroll
    for (int i = 0; i < 32; ++i) { v[i] = xr[lane + 32 * i]; s += v[i]; }
#pragma unroll
    for (int o = 16; o > 0; o >>= 1) s += __shfl_xor(s, o, 32);
    float mu = s * (1.0f / (float)D_MODEL);

    float vs = 0.0f;
#pragma unroll
    for (int i = 0; i < 32; ++i) { float d = v[i] - mu; vs += d * d; }
#pragma unroll
    for (int o = 16; o > 0; o >>= 1) vs += __shfl_xor(vs, o, 32);
    float inv = rsqrtf(vs * (1.0f / (float)D_MODEL) + 1e-5f);

    _Float16* dst = xn + (size_t)row * D_MODEL;
#pragma unroll
    for (int i = 0; i < 32; ++i) {
        int c = lane + 32 * i;
        dst[c] = (_Float16)((v[i] - mu) * inv * w[c] + b[c]);
    }
}

// ---------------------------------------------------------------------------
// WMMA GEMM body: C[M x N] = A[M x K] @ W[N x K]^T, block tile 128x128,
// 8 waves (4 M x 2 N), wave tile 32x64 (2x4 WMMA frags).
// A and W K-slices (128 x 32 f16) are double-buffered in LDS, filled with
// GLOBAL_LOAD_ASYNC_TO_LDS_B128 (ASYNCcnt) and read back with ds_load_b128.
// ---------------------------------------------------------------------------
template <int K, int N, bool F16OUT>
__device__ __forceinline__ void gemm_tile(
    const _Float16* __restrict__ A,
    const _Float16* __restrict__ W,
    const float*    __restrict__ bias,
    const float*    __restrict__ resid,   // only used when !F16OUT
    void*           __restrict__ outp)
{
    __shared__ _Float16 ldsA[2][128 * LDS_ROW];
    __shared__ _Float16 ldsB[2][128 * LDS_ROW];

    int tid  = threadIdx.x;
    int wave = tid >> 5;
    int lane = tid & 31;
    int wm = wave & 3, wn = wave >> 2;
    int row0  = blockIdx.y * 128 + wm * 32;
    int col0  = blockIdx.x * 128 + wn * 64;
    int brow0 = blockIdx.y * 128;
    int bcol0 = blockIdx.x * 128;

    // Fragment-read base offsets in LDS (halves).
    int m  = lane & 15;
    int uo = (lane >> 4) * 8;

    v8f acc[2][4] = {};

    // Stage one 128x32 K-slice of A and W into LDS buffer `buf`.
    // 1024 16-byte chunks total (512 A + 512 B); each thread moves 2+2.
    auto stage = [&](int buf, int k0) {
#pragma unroll
        for (int q = 0; q < 2; ++q) {
            int c    = tid * 2 + q;          // chunk id 0..511
            int crow = c >> 2;               // tile row
            int ckh  = (c & 3) * 8;          // k offset in halves
            _Float16* ga = const_cast<_Float16*>(
                A + (size_t)(brow0 + crow) * K + k0 + ckh);
            _Float16* gb = const_cast<_Float16*>(
                W + (size_t)(bcol0 + crow) * K + k0 + ckh);
            _Float16* la = &ldsA[buf][crow * LDS_ROW + ckh];
            _Float16* lb = &ldsB[buf][crow * LDS_ROW + ckh];
#if USE_ASYNC_LDS
            __builtin_amdgcn_global_load_async_to_lds_b128(
                (AS1 v4i*)ga, (AS3 v4i*)la, 0, 0);
            __builtin_amdgcn_global_load_async_to_lds_b128(
                (AS1 v4i*)gb, (AS3 v4i*)lb, 0, 0);
#else
            *(v8h*)la = *(const v8h*)ga;
            *(v8h*)lb = *(const v8h*)gb;
#endif
        }
    };

    constexpr int NK = K / 32;
    stage(0, 0);

    for (int s = 0; s < NK; ++s) {
        int cur = s & 1;
        if (s + 1 < NK) {
            stage(1 - cur, (s + 1) * 32);   // fill next buffer (async)
            wait_async<4>();                // this stage's 4 async loads done
        } else {
            wait_async<0>();
        }
        __syncthreads();                    // all waves' LDS writes visible

        v16h a[2], bf[4];
#pragma unroll
        for (int i = 0; i < 2; ++i)
            a[i] = load_frag_lds(&ldsA[cur][(wm * 32 + 16 * i + m) * LDS_ROW + uo]);
#pragma unroll
        for (int j = 0; j < 4; ++j)
            bf[j] = load_frag_lds(&ldsB[cur][(wn * 64 + 16 * j + m) * LDS_ROW + uo]);
#pragma unroll
        for (int i = 0; i < 2; ++i)
#pragma unroll
            for (int j = 0; j < 4; ++j)
                acc[i][j] = __builtin_amdgcn_wmma_f32_16x16x32_f16(
                    false, a[i], false, bf[j], (short)0, acc[i][j], false, false);

        __syncthreads();                    // done reading before refill
    }

    // Epilogue: C/D layout -> lane n = lane&15, M = VGPR idx (+8 upper lanes)
    int n  = lane & 15;
    int mo = (lane >> 4) * 8;
#pragma unroll
    for (int i = 0; i < 2; ++i) {
#pragma unroll
        for (int j = 0; j < 4; ++j) {
            int c = col0 + 16 * j + n;
            float bb = bias[c];
            int r = row0 + 16 * i + mo;
#pragma unroll
            for (int d = 0; d < 8; ++d) {
                size_t idx = (size_t)(r + d) * N + c;
                if constexpr (F16OUT) {
                    ((_Float16*)outp)[idx] = (_Float16)(acc[i][j][d] + bb);
                } else {
                    ((float*)outp)[idx] = acc[i][j][d] + bb + resid[idx];
                }
            }
        }
    }
}

// GEMM1: xz[M x 4096] = xn[M x 1024] @ in_w^T (+ in_b), f16 out
__global__ __launch_bounds__(256) void gemm_xz_kernel(
    const _Float16* __restrict__ A, const _Float16* __restrict__ W,
    const float* __restrict__ bias, _Float16* __restrict__ Cout) {
    gemm_tile<D_MODEL, 2 * D_INNER, true>(A, W, bias, nullptr, Cout);
}

// GEMM2: out[M x 1024] = y[M x 2048] @ out_w^T + out_b + residual, fp32 out
__global__ __launch_bounds__(256) void gemm_out_kernel(
    const _Float16* __restrict__ A, const _Float16* __restrict__ W,
    const float* __restrict__ bias, const float* __restrict__ resid,
    float* __restrict__ Cout) {
    gemm_tile<D_INNER, D_MODEL, false>(A, W, bias, resid, Cout);
}

// ---------------------------------------------------------------------------
// Fused depthwise conv (D_CONV=4, causal) + SiLU + gate dot + gating.
// One 256-thread block per row (b,t). y[M x 2048] f16.
// ---------------------------------------------------------------------------
__global__ __launch_bounds__(256) void conv_gate_kernel(
    const _Float16* __restrict__ xz,     // [NROWS x 4096]
    const float*    __restrict__ conv_w, // [2048 x 1 x 4]
    const float*    __restrict__ conv_b, // [2048]
    const float*    __restrict__ cw,     // [2048]
    const float*    __restrict__ Dp,     // [2048]
    _Float16*       __restrict__ y)      // [NROWS x 2048]
{
    const int N = 2 * D_INNER;
    int row = blockIdx.x;
    int t   = row & (SEQ - 1);
    int tid = threadIdx.x;

    __shared__ float red[256];

    float xcv[8];
    float partial = 0.0f;
#pragma unroll
    for (int it = 0; it < 8; ++it) {
        int c = tid + it * 256;
        float accv = conv_b[c];
#pragma unroll
        for (int j = 0; j < D_CONV; ++j) {
            int tt = t - (D_CONV - 1) + j;
            if (tt >= 0)
                accv += (float)xz[(size_t)(row - (D_CONV - 1) + j) * N + c] *
                        conv_w[c * D_CONV + j];
        }
        float sv = silu_f(accv);
        xcv[it] = sv;
        partial += sv * cw[c];
    }

    red[tid] = partial;
    __syncthreads();
    for (int s = 128; s > 0; s >>= 1) {
        if (tid < s) red[tid] += red[tid + s];
        __syncthreads();
    }
    float gate = 1.0f / (1.0f + __expf(-red[0]));

#pragma unroll
    for (int it = 0; it < 8; ++it) {
        int c = tid + it * 256;
        float zv = (float)xz[(size_t)row * N + D_INNER + c];
        y[(size_t)row * D_INNER + c] =
            (_Float16)(xcv[it] * (gate + Dp[c]) * silu_f(zv));
    }
}

// ---------------------------------------------------------------------------
// Launcher
// ---------------------------------------------------------------------------
extern "C" void kernel_launch(void* const* d_in, const int* in_sizes, int n_in,
                              void* d_out, int out_size, void* d_ws, size_t ws_size,
                              hipStream_t stream) {
    const float* x       = (const float*)d_in[0];
    const float* norm_w  = (const float*)d_in[1];
    const float* norm_b  = (const float*)d_in[2];
    const float* in_w    = (const float*)d_in[3];
    const float* in_b    = (const float*)d_in[4];
    const float* conv_w  = (const float*)d_in[5];
    const float* conv_b  = (const float*)d_in[6];
    const float* xproj_w = (const float*)d_in[7];
    // d_in[8] = A_log (unused by the reference computation path)
    const float* Dp      = (const float*)d_in[9];
    const float* out_w   = (const float*)d_in[10];
    const float* out_b   = (const float*)d_in[11];
    float* out = (float*)d_out;

    // Workspace layout
    char* ws = (char*)d_ws;
    _Float16* xn_f16 = (_Float16*)ws;  ws += (size_t)NROWS * D_MODEL * 2;         // 32 MB
    _Float16* w1_f16 = (_Float16*)ws;  ws += (size_t)(2 * D_INNER) * D_MODEL * 2; // 8 MB
    _Float16* w2_f16 = (_Float16*)ws;  ws += (size_t)D_MODEL * D_INNER * 2;       // 4 MB
    _Float16* xz_f16 = (_Float16*)ws;  ws += (size_t)NROWS * (2 * D_INNER) * 2;   // 128 MB
    _Float16* y_f16  = (_Float16*)ws;  ws += (size_t)NROWS * D_INNER * 2;         // 64 MB
    float*    cw     = (float*)ws;     ws += (size_t)D_INNER * 4;                 // 8 KB

    // 1) Weight precomputation
    {
        int n1 = (2 * D_INNER) * D_MODEL;  // 4,194,304
        cast_f32_to_f16_kernel<<<(n1 + 255) / 256, 256, 0, stream>>>(in_w, w1_f16, n1);
        int n2 = D_MODEL * D_INNER;        // 2,097,152
        cast_f32_to_f16_kernel<<<(n2 + 255) / 256, 256, 0, stream>>>(out_w, w2_f16, n2);
        cw_kernel<<<(D_INNER + 255) / 256, 256, 0, stream>>>(xproj_w, cw);
    }

    // 2) LayerNorm + cast
    ln_kernel<<<NROWS / 8, 256, 0, stream>>>(x, norm_w, norm_b, xn_f16);

    // 3) in_proj GEMM (WMMA + async LDS double buffer)
    {
        dim3 grid((2 * D_INNER) / 128, NROWS / 128);  // (32, 128)
        gemm_xz_kernel<<<grid, 256, 0, stream>>>(xn_f16, w1_f16, in_b, xz_f16);
    }

    // 4) Fused conv + SiLU + gate
    conv_gate_kernel<<<NROWS, 256, 0, stream>>>(xz_f16, conv_w, conv_b, cw, Dp, y_f16);

    // 5) out_proj GEMM (WMMA) + bias + residual
    {
        dim3 grid(D_MODEL / 128, NROWS / 128);        // (8, 128)
        gemm_out_kernel<<<grid, 256, 0, stream>>>(y_f16, w2_f16, out_b, x, out);
    }
}
